// MultiHeadAttentionRelative_54296976556831
// MI455X (gfx1250) — matrix operand order repaired and checked by reading, not compiled, not run on hardware
//
#include <hip/hip_runtime.h>
#include <hip/hip_bf16.h>
#include <hip/hip_fp16.h>

// ---------------------------------------------------------------------------
// MI455X (gfx1250) fused relative-position MHA.
// B=8, L=1024, D=64 (=head_dim), H=8, MAX_REL=150 (301-row tables).
// All GEMMs via v_wmma_f32_16x16x32_f16 (wave32); Q-tile staged into LDS via
// global_load_async_to_lds_b128 (ASYNCcnt).
// ---------------------------------------------------------------------------

typedef __attribute__((ext_vector_type(16))) _Float16 v16h;
typedef __attribute__((ext_vector_type(8)))  _Float16 v8h;
typedef __attribute__((ext_vector_type(8)))  float    v8f;

#define WMMA16(a, b, c) \
  __builtin_amdgcn_wmma_f32_16x16x32_f16(false, (a), false, (b), (short)0, (c), false, false)

#define BB   8
#define LL   1024
#define DD   64
#define HH   8
#define DM   512          // H*D
#define TQ   32           // queries per workgroup in attention kernel
#define NRP  304          // rel rows padded to x16 (301 -> 304)
#define NRK  320          // rel rows padded to x32 for WMMA K dim

// LDS layout byte offsets inside attn_kernel (dynamic LDS, no static LDS):
//   S   [TQ][LL]  f32 @ 0
//   P   [TQ][LL]  f16 @ 131072
//   R   [TQ][NRP] f32 @ 196608
//   W2  [TQ][NRK] f32 @ 235520
//   W2H [TQ][NRK] f16 @ 276480
//   QT  [TQ][DD]  f16 @ 296960      total 301056 B  (<= 320 KB / WGP)
#define QT_LDS_OFF 296960

// A-operand (16x32 f16, MxK): lane holds row m = lane&15; elems[0..15] =
// row[k0+hi*16 .. +15] (contiguous K, matching mk_b).
static __device__ inline v16h mk_a(const _Float16* rowptr, int k0, int hi) {
  return *(const v16h*)(rowptr + k0 + hi * 16);
}

// B-operand (32x16 f16, KxN): lane holds column n = lane&15, K group (lane>>4)
// -> 16 contiguous K values per lane = one 32B load.
static __device__ inline v16h mk_b(const _Float16* p) {
  return *(const v16h*)p;
}

// ---------------------------------------------------------------------------
// Kernel 1: dtype conversions + weight transposes + rel-table padding.
// ---------------------------------------------------------------------------
__global__ __launch_bounds__(256) void cvt_kernel(
    const float* __restrict__ q, const float* __restrict__ k, const float* __restrict__ v,
    const float* __restrict__ Wq, const float* __restrict__ Wk, const float* __restrict__ Wv,
    const float* __restrict__ Wo, const float* __restrict__ rk, const float* __restrict__ rv,
    _Float16* xq, _Float16* xk, _Float16* xv,
    _Float16* wqt, _Float16* wkt, _Float16* wvt, _Float16* wot,
    _Float16* relk, _Float16* relvt) {
  unsigned i = blockIdx.x * 256u + threadIdx.x;
  if (i < 524288u)  { xq[i] = (_Float16)q[i]; return; }
  if (i < 1048576u) { unsigned j = i - 524288u;  xk[j] = (_Float16)k[j]; return; }
  if (i < 1572864u) { unsigned j = i - 1048576u; xv[j] = (_Float16)v[j]; return; }
  if (i < 1605632u) { unsigned j = i - 1572864u; unsigned n = j >> 6, kk = j & 63u;
                      wqt[j] = (_Float16)Wq[kk * 512u + n]; return; }
  if (i < 1638400u) { unsigned j = i - 1605632u; unsigned n = j >> 6, kk = j & 63u;
                      wkt[j] = (_Float16)Wk[kk * 512u + n]; return; }
  if (i < 1671168u) { unsigned j = i - 1638400u; unsigned n = j >> 6, kk = j & 63u;
                      wvt[j] = (_Float16)Wv[kk * 512u + n]; return; }
  if (i < 1703936u) { unsigned j = i - 1671168u; unsigned n = j >> 9, kk = j & 511u;
                      wot[j] = (_Float16)Wo[kk * 64u + n]; return; }
  if (i < 1723392u) { unsigned j = i - 1703936u; unsigned r = j >> 6, d = j & 63u;
                      relk[j] = (r < 301u) ? (_Float16)rk[r * 64u + d] : (_Float16)0.f; return; }
  if (i < 1743872u) { unsigned j = i - 1723392u; unsigned d = j / 320u, r = j % 320u;
                      relvt[j] = (r < 301u) ? (_Float16)rv[r * 64u + d] : (_Float16)0.f; return; }
}

// ---------------------------------------------------------------------------
// Kernel 2: QKV projections. C[8192,512] = X[8192,64] @ W[64,512].
// grid = (2048, 1, 3); 8 waves/block, 1 16x16 tile per wave, 2 WMMA k-steps.
// V is stored transposed Vt[b][h][d][l]; per-lane the 8 C elements are 8
// consecutive l positions at fixed d -> one packed 16B store.
// ---------------------------------------------------------------------------
__global__ __launch_bounds__(256) void proj_kernel(
    const _Float16* __restrict__ xq, const _Float16* __restrict__ xk, const _Float16* __restrict__ xv,
    const _Float16* __restrict__ wqt, const _Float16* __restrict__ wkt, const _Float16* __restrict__ wvt,
    _Float16* __restrict__ Qf, _Float16* __restrict__ Kf, _Float16* __restrict__ Vt) {
  const int mode = blockIdx.z;
  const _Float16* X  = (mode == 0) ? xq  : (mode == 1) ? xk  : xv;
  const _Float16* Wt = (mode == 0) ? wqt : (mode == 1) ? wkt : wvt;
  const int tile = blockIdx.x * 8 + (threadIdx.x >> 5);
  const int lane = threadIdx.x & 31, lo = lane & 15, hi = lane >> 4;
  const int mt = tile >> 5, nt = tile & 31;     // 512 m-tiles x 32 n-tiles

  v8f acc = {};
#pragma unroll
  for (int ks = 0; ks < 2; ++ks) {
    const int k0 = ks * 32;
    v16h a = mk_a(X + (size_t)(mt * 16 + lo) * 64, k0, hi);
    v16h b = mk_b(Wt + (size_t)(nt * 16 + lo) * 64 + k0 + hi * 16);
    acc = WMMA16(a, b, acc);
  }
  if (mode < 2) {
    _Float16* dst = (mode == 0) ? Qf : Kf;
#pragma unroll
    for (int j = 0; j < 8; ++j)
      dst[(size_t)(mt * 16 + j + 8 * hi) * DM + nt * 16 + lo] = (_Float16)acc[j];
  } else {
    const int row0 = mt * 16 + 8 * hi;          // first of 8 consecutive rows
    const int col  = nt * 16 + lo;              // h*D + d
    const int bb = row0 >> 10, ll0 = row0 & 1023;
    const int hh = col >> 6,   dd  = col & 63;
    v8h pk = {};
#pragma unroll
    for (int j = 0; j < 8; ++j) pk[j] = (_Float16)acc[j];
    *(v8h*)(Vt + (((size_t)bb * HH + hh) * DD + dd) * LL + ll0) = pk;
  }
}

// ---------------------------------------------------------------------------
// Kernel 3: fused attention. One workgroup = one (b, h, 32-query tile).
// ---------------------------------------------------------------------------
__global__ __launch_bounds__(256) void attn_kernel(
    const _Float16* __restrict__ Qf, const _Float16* __restrict__ Kf,
    const _Float16* __restrict__ Vt, const _Float16* __restrict__ relk,
    const _Float16* __restrict__ relvt, _Float16* __restrict__ ctx) {
  extern __shared__ char smem[];
  float*    S   = (float*)smem;                       // [TQ][LL]
  _Float16* P   = (_Float16*)(S + TQ * LL);           // [TQ][LL]
  float*    R   = (float*)(P + TQ * LL);              // [TQ][NRP]
  float*    W2  = R + TQ * NRP;                       // [TQ][NRK]
  _Float16* W2H = (_Float16*)(W2 + TQ * NRK);         // [TQ][NRK]
  _Float16* QT  = W2H + TQ * NRK;                     // [TQ][DD] @ QT_LDS_OFF

  const int qt = blockIdx.x, h = blockIdx.y, b = blockIdx.z;
  const int tid = threadIdx.x;
  const int wave = tid >> 5, lane = tid & 31, lo = lane & 15, hi = lane >> 4;
  const int q0 = qt * TQ;

  // Async-copy the Q tile (4 KB = 256 lanes x 16 B) into LDS: lane t covers
  // QT halves [t*8 .. t*8+7]  ==  row m = t>>3, cols (t&7)*8 .. +7.
  {
    const int m  = tid >> 3, kk = (tid & 7) * 8;
    const _Float16* gsrc = Qf + (size_t)b * LL * DM + (size_t)(q0 + m) * DM +
                           (size_t)h * DD + kk;
    unsigned lds_off = QT_LDS_OFF + (unsigned)tid * 16u;   // byte offset in LDS
    unsigned long long gaddr = (unsigned long long)(uintptr_t)gsrc;
    asm volatile("global_load_async_to_lds_b128 %0, %1, off"
                 :: "v"(lds_off), "v"(gaddr)
                 : "memory");
  }
  // Overlap: zero the w2r scatter accumulator while the DMA runs.
  for (int i = tid; i < TQ * NRK; i += 256) W2[i] = 0.f;
  asm volatile("s_wait_asynccnt 0" ::: "memory");
  __syncthreads();

  // R[32][301] = Qtile @ rel_kᵀ   (2 x 19 tiles, K=64)
  for (int job = wave; job < 2 * 19; job += 8) {
    const int mt = job / 19, nt = job % 19;
    v8f acc = {};
#pragma unroll
    for (int ks = 0; ks < 2; ++ks) {
      const int k0 = ks * 32;
      v16h a = mk_a(QT + (mt * 16 + lo) * DD, k0, hi);
      v16h bm = mk_b(relk + (size_t)(nt * 16 + lo) * DD + k0 + hi * 16);
      acc = WMMA16(a, bm, acc);
    }
#pragma unroll
    for (int j = 0; j < 8; ++j)
      R[(mt * 16 + j + 8 * hi) * NRP + nt * 16 + lo] = acc[j];
  }
  // S[32][1024] = Qtile @ Kᵀ      (2 x 64 tiles, K=64)
  const _Float16* kbase = Kf + (size_t)b * LL * DM + (size_t)h * DD;
  for (int job = wave; job < 2 * 64; job += 8) {
    const int mt = job >> 6, nt = job & 63;
    v8f acc = {};
#pragma unroll
    for (int ks = 0; ks < 2; ++ks) {
      const int k0 = ks * 32;
      v16h a = mk_a(QT + (mt * 16 + lo) * DD, k0, hi);
      v16h bm = mk_b(kbase + (size_t)(nt * 16 + lo) * DM + k0 + hi * 16);
      acc = WMMA16(a, bm, acc);
    }
#pragma unroll
    for (int j = 0; j < 8; ++j)
      S[(mt * 16 + j + 8 * hi) * LL + nt * 16 + lo] = acc[j];
  }
  __syncthreads();

  // --- softmax over 1024 keys: 8 lanes per row, 128 cols each -------------
  const int row = tid >> 3, sub = tid & 7;
  const int qg = q0 + row;
  float* srow = S + row * LL;
  float* rrow = R + row * NRP;
  const float inv_scale = 0.125f;  // 1/sqrt(64)

  float mx = -3.0e38f;
  for (int i = 0; i < 128; ++i) {
    const int kk = sub * 128 + i;
    int d = kk - qg; d = d < -150 ? -150 : (d > 150 ? 150 : d);
    const float z = (srow[kk] + rrow[d + 150]) * inv_scale;
    srow[kk] = z;
    mx = fmaxf(mx, z);
  }
  mx = fmaxf(mx, __shfl_xor(mx, 1));
  mx = fmaxf(mx, __shfl_xor(mx, 2));
  mx = fmaxf(mx, __shfl_xor(mx, 4));

  float sum = 0.f;
  for (int i = 0; i < 128; ++i) {
    const int kk = sub * 128 + i;
    const float e = __expf(srow[kk] - mx);
    srow[kk] = e;
    sum += e;
  }
  sum += __shfl_xor(sum, 1);
  sum += __shfl_xor(sum, 2);
  sum += __shfl_xor(sum, 4);
  const float inv = 1.0f / sum;

  _Float16* prow = P + row * LL;
  float* wrow = W2 + row * NRK;
  for (int i = 0; i < 128; ++i) {
    const int kk = sub * 128 + i;
    const float p = srow[kk] * inv;
    prow[kk] = (_Float16)p;
    int d = kk - qg; d = d < -150 ? -150 : (d > 150 ? 150 : d);
    atomicAdd(&wrow[d + 150], p);        // ds_add_f32 scatter into w2r
  }
  __syncthreads();

  for (int i = tid; i < TQ * NRK; i += 256) W2H[i] = (_Float16)W2[i];
  __syncthreads();

  // --- ctx = P @ V  +  w2r @ rel_v : one 16x16 tile per wave --------------
  {
    const int mt = wave >> 2, nt = wave & 3;
    v8f acc = {};
    const _Float16* vb = Vt + ((size_t)b * HH + h) * DD * LL;  // [d][l]
    for (int ks = 0; ks < 32; ++ks) {
      const int k0 = ks * 32;
      __builtin_prefetch(vb + (size_t)(nt * 16 + lo) * LL + k0 + 256, 0, 1);
      v16h a = mk_a(P + (mt * 16 + lo) * LL, k0, hi);
      v16h bm = mk_b(vb + (size_t)(nt * 16 + lo) * LL + k0 + hi * 16);
      acc = WMMA16(a, bm, acc);
    }
#pragma unroll
    for (int ks = 0; ks < 10; ++ks) {
      const int k0 = ks * 32;
      v16h a = mk_a(W2H + (mt * 16 + lo) * NRK, k0, hi);
      v16h bm = mk_b(relvt + (size_t)(nt * 16 + lo) * NRK + k0 + hi * 16);
      acc = WMMA16(a, bm, acc);
    }
    _Float16* cb = ctx + (size_t)b * LL * DM + (size_t)h * DD;
#pragma unroll
    for (int j = 0; j < 8; ++j)
      cb[(size_t)(q0 + mt * 16 + j + 8 * hi) * DM + nt * 16 + lo] = (_Float16)acc[j];
  }
}

// ---------------------------------------------------------------------------
// Kernel 4: out = ctx @ Wo.  [8192,512]x[512,64] -> f32 output.
// ---------------------------------------------------------------------------
__global__ __launch_bounds__(256) void out_kernel(
    const _Float16* __restrict__ ctx, const _Float16* __restrict__ wot,
    float* __restrict__ out) {
  const int tile = blockIdx.x * 8 + (threadIdx.x >> 5);
  const int lane = threadIdx.x & 31, lo = lane & 15, hi = lane >> 4;
  const int mt = tile >> 2, nt = tile & 3;
  v8f acc = {};
  for (int ks = 0; ks < 16; ++ks) {
    const int k0 = ks * 32;
    v16h a = mk_a(ctx + (size_t)(mt * 16 + lo) * DM, k0, hi);
    v16h b = mk_b(wot + (size_t)(nt * 16 + lo) * DM + k0 + hi * 16);
    acc = WMMA16(a, b, acc);
  }
#pragma unroll
  for (int j = 0; j < 8; ++j)
    out[(size_t)(mt * 16 + j + 8 * hi) * DD + nt * 16 + lo] = acc[j];
}

// ---------------------------------------------------------------------------
extern "C" void kernel_launch(void* const* d_in, const int* in_sizes, int n_in,
                              void* d_out, int out_size, void* d_ws, size_t ws_size,
                              hipStream_t stream) {
  const float* q  = (const float*)d_in[0];
  const float* k  = (const float*)d_in[1];
  const float* v  = (const float*)d_in[2];
  const float* Wq = (const float*)d_in[3];
  const float* Wk = (const float*)d_in[4];
  const float* Wv = (const float*)d_in[5];
  const float* Wo = (const float*)d_in[6];
  const float* rk = (const float*)d_in[7];
  const float* rv = (const float*)d_in[8];
  float* out = (float*)d_out;

  char* ws = (char*)d_ws;
  _Float16* xq    = (_Float16*)(ws + 0);
  _Float16* xk    = (_Float16*)(ws + 1048576);
  _Float16* xv    = (_Float16*)(ws + 2097152);
  _Float16* wqt   = (_Float16*)(ws + 3145728);
  _Float16* wkt   = (_Float16*)(ws + 3211264);
  _Float16* wvt   = (_Float16*)(ws + 3276800);
  _Float16* wot   = (_Float16*)(ws + 3342336);
  _Float16* relk  = (_Float16*)(ws + 3407872);
  _Float16* relvt = (_Float16*)(ws + 3446784);
  _Float16* Qf    = (_Float16*)(ws + 3487744);
  _Float16* Kf    = (_Float16*)(ws + 11876352);
  _Float16* Vt    = (_Float16*)(ws + 20264960);
  _Float16* ctx   = (_Float16*)(ws + 28653568);
  // total ws use: 37,042,176 bytes

  cvt_kernel<<<6812, 256, 0, stream>>>(q, k, v, Wq, Wk, Wv, Wo, rk, rv,
                                       xq, xk, xv, wqt, wkt, wvt, wot, relk, relvt);
  proj_kernel<<<dim3(2048, 1, 3), 256, 0, stream>>>(xq, xk, xv, wqt, wkt, wvt,
                                                    Qf, Kf, Vt);
  const size_t lds_bytes =
      (size_t)TQ * LL * 4 + (size_t)TQ * LL * 2 + (size_t)TQ * NRP * 4 +
      (size_t)TQ * NRK * 4 + (size_t)TQ * NRK * 2 + (size_t)TQ * DD * 2;  // 301,056
  attn_kernel<<<dim3(LL / TQ, HH, BB), 256, lds_bytes, stream>>>(Qf, Kf, Vt, relk,
                                                                 relvt, ctx);
  out_kernel<<<256, 256, 0, stream>>>(ctx, wot, out);
}
